// TransformerBlock_8495445311961
// MI455X (gfx1250) — matrix operand order repaired and checked by reading, not compiled, not run on hardware
//
#include <hip/hip_runtime.h>

typedef __attribute__((ext_vector_type(16))) __bf16 v16bf;
typedef __attribute__((ext_vector_type(8)))  float  v8f;

constexpr int Bc = 4, Tc = 512, Dc = 768, Hc = 12, FFc = 3072, TBc = 64, DKc = 64;
constexpr int NTc = Bc * Tc; // 2048 rows

#if defined(__HIP_DEVICE_COMPILE__) && __has_builtin(__builtin_amdgcn_global_load_async_to_lds_b128)
#define HAVE_ASYNC 1
#else
#define HAVE_ASYNC 0
#endif

typedef int v4i_t __attribute__((vector_size(16)));
typedef __attribute__((address_space(1))) v4i_t as1_v4i;
typedef __attribute__((address_space(3))) v4i_t as3_v4i;

#if HAVE_ASYNC
__device__ __forceinline__ void async_b128(const void* g, void* l) {
  __builtin_amdgcn_global_load_async_to_lds_b128(
      (as1_v4i*)(unsigned long long)g,
      (as3_v4i*)(unsigned)(unsigned long long)l, 0, 0);
}
__device__ __forceinline__ void wait_async0() {
#if __has_builtin(__builtin_amdgcn_s_wait_asynccnt)
  __builtin_amdgcn_s_wait_asynccnt(0);
#else
  asm volatile("s_wait_asynccnt 0x0" ::: "memory");
#endif
}
#endif

__device__ __forceinline__ unsigned short f2bf(float f) {
  union { float f; unsigned u; } v; v.f = f;
  unsigned u = v.u;
  return (unsigned short)((u + 0x7FFFu + ((u >> 16) & 1u)) >> 16); // RNE
}

// ---------------- f32 -> bf16 transpose-convert: src[K][N] -> dst[N][K] ----------------
__global__ __launch_bounds__(256) void cvt_bf16_t_kernel(const float* __restrict__ src,
                                                         unsigned short* __restrict__ dst,
                                                         int K, int N) {
  __shared__ float tile[32][33];
  int nb = blockIdx.x * 32, kb = blockIdx.y * 32;
  int tx = threadIdx.x & 31, ty = threadIdx.x >> 5; // 32 x 8
  for (int i = ty; i < 32; i += 8)
    tile[i][tx] = src[(size_t)(kb + i) * N + nb + tx];
  __syncthreads();
  for (int i = ty; i < 32; i += 8)
    dst[(size_t)(nb + i) * K + kb + tx] = f2bf(tile[tx][i]);
}

// ---------------- LayerNorm (one row per block) ----------------
__global__ __launch_bounds__(256) void ln_kernel(const float* __restrict__ x,
                                                 const float* __restrict__ g,
                                                 const float* __restrict__ bta,
                                                 unsigned short* __restrict__ out) {
  __shared__ float red[256];
  size_t row = blockIdx.x;
  const float* xr = x + row * Dc;
  unsigned short* orow = out + row * Dc;
  int tid = threadIdx.x;
  float s = 0.f;
  for (int i = tid; i < Dc; i += 256) s += xr[i];
  red[tid] = s; __syncthreads();
  for (int st = 128; st > 0; st >>= 1) { if (tid < st) red[tid] += red[tid + st]; __syncthreads(); }
  float mean = red[0] / (float)Dc; __syncthreads();
  float s2 = 0.f;
  for (int i = tid; i < Dc; i += 256) { float d = xr[i] - mean; s2 += d * d; }
  red[tid] = s2; __syncthreads();
  for (int st = 128; st > 0; st >>= 1) { if (tid < st) red[tid] += red[tid + st]; __syncthreads(); }
  float inv = rsqrtf(red[0] / (float)Dc + 1e-5f);
  for (int i = tid; i < Dc; i += 256)
    orow[i] = f2bf((xr[i] - mean) * inv * g[i] + bta[i]);
}

// ---------------- temporal-bias MLP ----------------
__global__ __launch_bounds__(256) void bias_mlp_kernel(const float* __restrict__ dt,
    const float* __restrict__ w1, const float* __restrict__ b1,
    const float* __restrict__ w2, const float* __restrict__ b2,
    float* __restrict__ outb, size_t n) {
  __shared__ float sw1[TBc], sb1[TBc], sw2[TBc];
  int tid = threadIdx.x;
  if (tid < TBc) { sw1[tid] = w1[tid]; sb1[tid] = b1[tid]; sw2[tid] = w2[tid]; }
  __syncthreads();
  size_t i = (size_t)blockIdx.x * 256 + tid;
  if (i >= n) return;
  float tm = 1.0f / logf(2.718281828459045f + dt[i]);
  float acc = b2[0];
#pragma unroll
  for (int k = 0; k < TBc; k++) {
    float h = tm * sw1[k] + sb1[k];
    h = h > 0.f ? h : 0.2f * h;
    acc += h * sw2[k];
  }
  outb[i] = acc;
}

// ---------------- masked softmax over row of T, writes bf16 P ----------------
__global__ __launch_bounds__(256) void softmax_kernel(const float* __restrict__ scores,
    const float* __restrict__ biasM, const int* __restrict__ pad,
    unsigned short* __restrict__ P) {
  __shared__ float red[256];
  int i = blockIdx.x, h = blockIdx.y, b = blockIdx.z, tid = threadIdx.x;
  const float* srow = scores + (((size_t)(b * Hc + h)) * Tc + i) * Tc;
  const float* brow = biasM + (((size_t)b) * Tc + i) * Tc;
  const int* prow = pad + (((size_t)b) * Tc + i) * Tc;
  unsigned short* orow = P + (((size_t)(b * Hc + h)) * Tc + i) * Tc;
  float v[2];
#pragma unroll
  for (int jj = 0; jj < 2; jj++) {
    int j = tid + 256 * jj;
    float s = srow[j] + brow[j];
    if (prow[j] == 0) s = -1e9f;
    if (j > i && i != 0) s = -1e9f;   // causal, CLS row fully visible
    v[jj] = s;
  }
  float m = fmaxf(v[0], v[1]);
  red[tid] = m; __syncthreads();
  for (int st = 128; st > 0; st >>= 1) { if (tid < st) red[tid] = fmaxf(red[tid], red[tid + st]); __syncthreads(); }
  m = red[0]; __syncthreads();
  float e0 = __expf(v[0] - m), e1 = __expf(v[1] - m);
  red[tid] = e0 + e1; __syncthreads();
  for (int st = 128; st > 0; st >>= 1) { if (tid < st) red[tid] += red[tid + st]; __syncthreads(); }
  float inv = 1.0f / red[0];
  orow[tid] = f2bf(e0 * inv);
  orow[tid + 256] = f2bf(e1 * inv);
}

// ---------------- WMMA bf16 GEMM: BM=128 BN=64 BK=32, 8 waves / block ----------------
// B operand is always pre-transposed: global B is [N][K] row-major.
enum { EPI_SCALE_F32 = 0, EPI_BIAS_BF16 = 1, EPI_BIAS_GELU_BF16 = 2,
       EPI_BIAS_RES_F32 = 3, EPI_BF16 = 4, EPI_BIAS_BF16_TV = 5 };

template <int EPI>
__global__ __launch_bounds__(256) void gemm_wmma_kernel(
    const unsigned short* __restrict__ A, int lda, long long aSB, long long aSH,
    const unsigned short* __restrict__ Bm, int ldb, long long bSB, long long bSH,
    void* __restrict__ Cv, int ldc, long long cSB, long long cSH,
    const float* __restrict__ bias, const float* __restrict__ res,
    int K, int hSplit, float alpha) {
  constexpr int BM = 128, BN = 64, BK = 32;
  constexpr int LP = 40; // 80B row stride: 16B-aligned for async B128, conflict-free frag reads
  __shared__ unsigned short sA[2][BM * LP];
  __shared__ unsigned short sB[2][BN * LP];

  int tid = threadIdx.x, lane = tid & 31, wid = tid >> 5;
  int half = lane >> 4, ml = lane & 15;
  int waveM = wid & 3, waveN = wid >> 2;
  int rowBase = blockIdx.y * BM, colBase = blockIdx.x * BN;
  int z = blockIdx.z; int bb = z / hSplit, hh = z - bb * hSplit;

  A  += (size_t)bb * aSB + (size_t)hh * aSH;
  Bm += (size_t)bb * bSB + (size_t)hh * bSH;
  size_t coff = (size_t)bb * cSB + (size_t)hh * cSH;

  // per-thread staging coordinates (16B chunks)
  int ar0 = (tid) >> 2,        ac0 = (tid & 3) * 8;         // A chunk 0
  int ar1 = (tid + 256) >> 2,  ac1 = ac0;                   // A chunk 1
  int br  = tid >> 2,          bc  = (tid & 3) * 8;         // B chunk

  v8f zero = {};
  v8f acc[2][2] = {{zero, zero}, {zero, zero}};

  int nk = K / BK;

#if HAVE_ASYNC
  auto issue_tile = [&](int kb, int buf) {
    async_b128(A + (size_t)(rowBase + ar0) * lda + kb + ac0, &sA[buf][ar0 * LP + ac0]);
    async_b128(A + (size_t)(rowBase + ar1) * lda + kb + ac1, &sA[buf][ar1 * LP + ac1]);
    async_b128(Bm + (size_t)(colBase + br) * ldb + kb + bc,  &sB[buf][br * LP + bc]);
  };
  issue_tile(0, 0);
  wait_async0();
  __syncthreads();
#endif

  for (int i = 0; i < nk; i++) {
    int cur = i & 1;
#if HAVE_ASYNC
    if (i + 1 < nk) issue_tile((i + 1) * BK, cur ^ 1);
#else
    cur = 0;
    {
      int kb = i * BK;
      uint4 d0 = *(const uint4*)(A + (size_t)(rowBase + ar0) * lda + kb + ac0);
      uint4 d1 = *(const uint4*)(A + (size_t)(rowBase + ar1) * lda + kb + ac1);
      uint4 db = *(const uint4*)(Bm + (size_t)(colBase + br) * ldb + kb + bc);
      unsigned* w0 = (unsigned*)(&sA[0][ar0 * LP + ac0]);
      w0[0] = d0.x; w0[1] = d0.y; w0[2] = d0.z; w0[3] = d0.w;
      unsigned* w1 = (unsigned*)(&sA[0][ar1 * LP + ac1]);
      w1[0] = d1.x; w1[1] = d1.y; w1[2] = d1.z; w1[3] = d1.w;
      unsigned* w2 = (unsigned*)(&sB[0][br * LP + bc]);
      w2[0] = db.x; w2[1] = db.y; w2[2] = db.z; w2[3] = db.w;
    }
    __syncthreads();
#endif

    // fragments per documented 16-bit A 16x32 / B 32x16 VGPR layouts
    v16bf afr[2], bfr[2];
#pragma unroll
    for (int tm = 0; tm < 2; tm++) {
      int mrow = waveM * 32 + tm * 16 + ml;
      union { unsigned u[8]; v16bf v; } ua;
#pragma unroll
      for (int r = 0; r < 8; r++) {
        int k0 = 2 * r + 8 * half + ((r >= 4) ? 8 : 0);
        ua.u[r] = *(const unsigned*)(&sA[cur][mrow * LP + k0]);
      }
      afr[tm] = ua.v;
    }
#pragma unroll
    for (int tn = 0; tn < 2; tn++) {
      int ncol = waveN * 32 + tn * 16 + ml;
      union { unsigned u[8]; v16bf v; } ub;
#pragma unroll
      for (int r = 0; r < 8; r++) {
        int k0 = 2 * r + 8 * half + ((r >= 4) ? 8 : 0);
        ub.u[r] = *(const unsigned*)(&sB[cur][ncol * LP + k0]);
      }
      bfr[tn] = ub.v;
    }
#pragma unroll
    for (int tm = 0; tm < 2; tm++)
#pragma unroll
      for (int tn = 0; tn < 2; tn++)
        acc[tm][tn] = __builtin_amdgcn_wmma_f32_16x16x32_bf16(
            false, afr[tm], false, bfr[tn], (short)0, acc[tm][tn], false, false);

#if HAVE_ASYNC
    if (i + 1 < nk) { wait_async0(); __syncthreads(); }
#else
    __syncthreads();
#endif
  }

  // epilogue (C layout: vgpr r -> M=r+8*half, lane&15 -> N)
#pragma unroll
  for (int tm = 0; tm < 2; tm++) {
#pragma unroll
    for (int tn = 0; tn < 2; tn++) {
#pragma unroll
      for (int r = 0; r < 8; r++) {
        int m = rowBase + waveM * 32 + tm * 16 + r + 8 * half;
        int n = colBase + waveN * 32 + tn * 16 + ml;
        size_t ci = (size_t)m * ldc + n;
        float v = acc[tm][tn][r];
        if (EPI == EPI_SCALE_F32) {
          ((float*)Cv + coff)[ci] = v * alpha;
        } else if (EPI == EPI_BIAS_BF16) {
          ((unsigned short*)Cv + coff)[ci] = f2bf(v + bias[n]);
        } else if (EPI == EPI_BIAS_GELU_BF16) {
          float xg = v + bias[n];
          float gg = 0.5f * xg * (1.0f + erff(xg * 0.70710678118654752f));
          ((unsigned short*)Cv + coff)[ci] = f2bf(gg);
        } else if (EPI == EPI_BIAS_RES_F32) {
          ((float*)Cv + coff)[ci] = v + bias[n] + (res + coff)[ci];
        } else if (EPI == EPI_BIAS_BF16_TV) {
          // write V transposed per batch: Vt[b][n][t], m = b*T + t
          int bidx = m >> 9, t = m & (Tc - 1);
          ((unsigned short*)Cv)[(size_t)bidx * (Dc * Tc) + (size_t)n * Tc + t] = f2bf(v + bias[n]);
        } else {
          ((unsigned short*)Cv + coff)[ci] = f2bf(v);
        }
      }
    }
  }
}

extern "C" void kernel_launch(void* const* d_in, const int* in_sizes, int n_in,
                              void* d_out, int out_size, void* d_ws, size_t ws_size,
                              hipStream_t stream) {
  (void)in_sizes; (void)n_in; (void)out_size; (void)ws_size;
  const float* x    = (const float*)d_in[0];
  const float* dt   = (const float*)d_in[1];
  const int*   pad  = (const int*)d_in[2];
  const float* wq   = (const float*)d_in[3];  const float* bq   = (const float*)d_in[4];
  const float* wk   = (const float*)d_in[5];  const float* bk   = (const float*)d_in[6];
  const float* wv   = (const float*)d_in[7];  const float* bv   = (const float*)d_in[8];
  const float* wp   = (const float*)d_in[9];  const float* bp   = (const float*)d_in[10];
  const float* tb1w = (const float*)d_in[11]; const float* tb1b = (const float*)d_in[12];
  const float* tb2w = (const float*)d_in[13]; const float* tb2b = (const float*)d_in[14];
  const float* fc1w = (const float*)d_in[15]; const float* fc1b = (const float*)d_in[16];
  const float* fc2w = (const float*)d_in[17]; const float* fc2b = (const float*)d_in[18];
  const float* ln1g = (const float*)d_in[19]; const float* ln1b = (const float*)d_in[20];
  const float* ln2g = (const float*)d_in[21]; const float* ln2b = (const float*)d_in[22];

  char* base = (char*)d_ws; size_t off = 0;
  auto alloc = [&](size_t bytes) -> void* {
    void* p = base + off; off = (off + bytes + 255) & ~(size_t)255; return p;
  };

  unsigned short* h1    = (unsigned short*)alloc((size_t)NTc * Dc * 2);
  unsigned short* wqT   = (unsigned short*)alloc((size_t)Dc * Dc * 2);
  unsigned short* wkT   = (unsigned short*)alloc((size_t)Dc * Dc * 2);
  unsigned short* wvT   = (unsigned short*)alloc((size_t)Dc * Dc * 2);
  unsigned short* wpT   = (unsigned short*)alloc((size_t)Dc * Dc * 2);
  unsigned short* fc1T  = (unsigned short*)alloc((size_t)Dc * FFc * 2);
  unsigned short* fc2T  = (unsigned short*)alloc((size_t)FFc * Dc * 2);
  unsigned short* qb    = (unsigned short*)alloc((size_t)NTc * Dc * 2);
  unsigned short* kbm   = (unsigned short*)alloc((size_t)NTc * Dc * 2);
  unsigned short* vt    = (unsigned short*)alloc((size_t)NTc * Dc * 2); // V transposed per batch
  float*          biasM = (float*)alloc((size_t)Bc * Tc * Tc * 4);
  float*          sc    = (float*)alloc((size_t)Bc * Hc * Tc * Tc * 4);
  unsigned short* Pm    = (unsigned short*)alloc((size_t)Bc * Hc * Tc * Tc * 2);
  unsigned short* ctx   = (unsigned short*)alloc((size_t)NTc * Dc * 2);
  float*          x2    = (float*)alloc((size_t)NTc * Dc * 4);
  unsigned short* h2    = (unsigned short*)alloc((size_t)NTc * Dc * 2);
  unsigned short* mid   = (unsigned short*)alloc((size_t)NTc * FFc * 2);

  // transpose-convert all weights to bf16 [N][K]
  cvt_bf16_t_kernel<<<dim3(Dc / 32, Dc / 32), 256, 0, stream>>>(wq, wqT, Dc, Dc);
  cvt_bf16_t_kernel<<<dim3(Dc / 32, Dc / 32), 256, 0, stream>>>(wk, wkT, Dc, Dc);
  cvt_bf16_t_kernel<<<dim3(Dc / 32, Dc / 32), 256, 0, stream>>>(wv, wvT, Dc, Dc);
  cvt_bf16_t_kernel<<<dim3(Dc / 32, Dc / 32), 256, 0, stream>>>(wp, wpT, Dc, Dc);
  cvt_bf16_t_kernel<<<dim3(FFc / 32, Dc / 32), 256, 0, stream>>>(fc1w, fc1T, Dc, FFc);
  cvt_bf16_t_kernel<<<dim3(Dc / 32, FFc / 32), 256, 0, stream>>>(fc2w, fc2T, FFc, Dc);

  // LN1
  ln_kernel<<<NTc, 256, 0, stream>>>(x, ln1g, ln1b, h1);

  dim3 blk(256);
  // Q, K = h @ W + b (bf16 [N][D]); V written transposed per head/batch
  gemm_wmma_kernel<EPI_BIAS_BF16><<<dim3(Dc / 64, NTc / 128, 1), blk, 0, stream>>>(
      h1, Dc, 0, 0, wqT, Dc, 0, 0, qb, Dc, 0, 0, bq, nullptr, Dc, 1, 1.0f);
  gemm_wmma_kernel<EPI_BIAS_BF16><<<dim3(Dc / 64, NTc / 128, 1), blk, 0, stream>>>(
      h1, Dc, 0, 0, wkT, Dc, 0, 0, kbm, Dc, 0, 0, bk, nullptr, Dc, 1, 1.0f);
  gemm_wmma_kernel<EPI_BIAS_BF16_TV><<<dim3(Dc / 64, NTc / 128, 1), blk, 0, stream>>>(
      h1, Dc, 0, 0, wvT, Dc, 0, 0, vt, Tc, 0, 0, bv, nullptr, Dc, 1, 1.0f);

  // temporal bias MLP
  bias_mlp_kernel<<<dim3((Bc * Tc * Tc + 255) / 256), 256, 0, stream>>>(
      dt, tb1w, tb1b, tb2w, tb2b, biasM, (size_t)Bc * Tc * Tc);

  // scores = (Q @ K^T) / sqrt(DK)  (batched over B*H; K-matrix is naturally [N][K])
  gemm_wmma_kernel<EPI_SCALE_F32><<<dim3(Tc / 64, Tc / 128, Bc * Hc), blk, 0, stream>>>(
      qb, Dc, (long long)Tc * Dc, DKc,
      kbm, Dc, (long long)Tc * Dc, DKc,
      sc, Tc, (long long)Hc * Tc * Tc, (long long)Tc * Tc,
      nullptr, nullptr, DKc, Hc, 0.125f);

  // softmax (bias + masks fused) -> bf16 P
  softmax_kernel<<<dim3(Tc, Hc, Bc), 256, 0, stream>>>(sc, biasM, pad, Pm);

  // ctx = P @ V  (B operand = Vt, already [N=dk][K=t])
  gemm_wmma_kernel<EPI_BF16><<<dim3(DKc / 64, Tc / 128, Bc * Hc), blk, 0, stream>>>(
      Pm, Tc, (long long)Hc * Tc * Tc, (long long)Tc * Tc,
      vt, Tc, (long long)Dc * Tc, (long long)DKc * Tc,
      ctx, Dc, (long long)Tc * Dc, DKc,
      nullptr, nullptr, Tc, Hc, 1.0f);

  // x2 = x + ctx @ Wp + bp
  gemm_wmma_kernel<EPI_BIAS_RES_F32><<<dim3(Dc / 64, NTc / 128, 1), blk, 0, stream>>>(
      ctx, Dc, 0, 0, wpT, Dc, 0, 0, x2, Dc, 0, 0, bp, x, Dc, 1, 1.0f);

  // LN2
  ln_kernel<<<NTc, 256, 0, stream>>>(x2, ln2g, ln2b, h2);

  // mid = gelu(h2 @ fc1 + b1) (bf16)
  gemm_wmma_kernel<EPI_BIAS_GELU_BF16><<<dim3(FFc / 64, NTc / 128, 1), blk, 0, stream>>>(
      h2, Dc, 0, 0, fc1T, Dc, 0, 0, mid, FFc, 0, 0, fc1b, nullptr, Dc, 1, 1.0f);

  // out = x2 + mid @ fc2 + b2 (f32, directly to d_out)
  gemm_wmma_kernel<EPI_BIAS_RES_F32><<<dim3(Dc / 64, NTc / 128, 1), blk, 0, stream>>>(
      mid, FFc, 0, 0, fc2T, FFc, 0, 0, (float*)d_out, Dc, 0, 0, fc2b, x2, FFc, 1, 1.0f);
}
